// StraightThroughQuantizer_40518721470848
// MI455X (gfx1250) — matrix operand order repaired and checked by reading, not compiled, not run on hardware
//
#include <hip/hip_runtime.h>

// ---------------------------------------------------------------------------
// VQ straight-through quantizer for MI455X (gfx1250, wave32, WMMA).
//   z: [16,2048,128] f32   codebook: [4096,128] f32
// Outputs (concat, f32): quantized_z [N*D], indices [N], loss, commit, quant.
// ---------------------------------------------------------------------------

typedef __attribute__((ext_vector_type(16))) __bf16 v16bf;
typedef __attribute__((ext_vector_type(8)))  __bf16 v8bf;
typedef __attribute__((ext_vector_type(8)))  float  v8f;

#define NROWS (16 * 2048)   // 32768
#define DDIM  128
#define KCODE 4096
#define KT    64            // codes staged per LDS tile
#define ROWB  272           // LDS bytes per code row: 256 data + 16 pad (16B aligned, spreads banks)

__device__ __forceinline__ unsigned short f2bf_u16(float f) {
    union { float f; unsigned u; } a; a.f = f;
    unsigned r = a.u + 0x7FFFu + ((a.u >> 16) & 1u);   // RNE
    return (unsigned short)(r >> 16);
}
__device__ __forceinline__ __bf16 u16_bf(unsigned short s) {
    union { unsigned short s; __bf16 b; } o; o.s = s; return o.b;
}
__device__ __forceinline__ float bf_f(unsigned short s) {
    union { unsigned u; float f; } o; o.u = ((unsigned)s) << 16; return o.f;
}

__device__ __forceinline__ v8f wmma_bf16(v16bf a, v16bf b, v8f c) {
    // D = A(16x32 bf16) x B(32x16 bf16) + C(16x16 f32)
    return __builtin_amdgcn_wmma_f32_16x16x32_bf16(
        /*neg_a=*/false, a, /*neg_b=*/false, b,
        /*c_mod=*/(short)0, c, /*reuse_a=*/false, /*reuse_b=*/false);
}

// ---------------------------------------------------------------------------
// Kernel 1: split codebook into bf16 hi/lo planes + exact fp32 norms.
// 8 waves per block, one code row per wave.
// ---------------------------------------------------------------------------
__global__ __launch_bounds__(256) void vq_cb_prep(const float* __restrict__ cb,
                                                  unsigned short* __restrict__ hi,
                                                  unsigned short* __restrict__ lo,
                                                  float* __restrict__ nrm) {
    const int wave = threadIdx.x >> 5, lane = threadIdx.x & 31;
    const int row  = blockIdx.x * 8 + wave;
    const float* src = cb + (size_t)row * DDIM;
    float s = 0.f;
#pragma unroll
    for (int j = 0; j < 4; ++j) {
        const int d = lane + 32 * j;        // coalesced
        float f = src[d];
        s = fmaf(f, f, s);
        unsigned short hb = f2bf_u16(f);
        hi[(size_t)row * DDIM + d] = hb;
        lo[(size_t)row * DDIM + d] = f2bf_u16(f - bf_f(hb));
    }
#pragma unroll
    for (int off = 16; off; off >>= 1) s += __shfl_xor(s, off, 32);
    if (lane == 0) nrm[row] = s;
}

// ---------------------------------------------------------------------------
// Kernel 2: fused GEMM + argmin.
// Block = 8 waves x 32; wave handles 16 z-rows vs all 4096 codes.
// A (z tile, hi+lo) held in registers; codebook tiles staged via LDS.
// dist = ||w||^2 - 2 z.w   (||z||^2 is row-constant -> irrelevant to argmin)
// ---------------------------------------------------------------------------
__global__ __launch_bounds__(256) void vq_argmin(const float* __restrict__ z,
                                                 const unsigned short* __restrict__ cbhi,
                                                 const unsigned short* __restrict__ cblo,
                                                 const float* __restrict__ cbnorm,
                                                 int* __restrict__ out_idx) {
    __shared__ unsigned char ldsHi[KT * ROWB];
    __shared__ unsigned char ldsLo[KT * ROWB];

    const int wave = threadIdx.x >> 5;
    const int lane = threadIdx.x & 31;
    const int h    = lane >> 4;      // half-wave
    const int nm   = lane & 15;      // A: row m; B: column n
    const int row_base = blockIdx.x * 128 + wave * 16;

    // --- Build A operands: lane(h,m), elem e ->
    //     K = 16h + e, for k-step s covering dims 32s..32s+31.
    v16bf Ah[4], Al[4];
    {
        const float* zr = z + (size_t)(row_base + nm) * DDIM;
#pragma unroll
        for (int s = 0; s < 4; ++s) {
#pragma unroll
            for (int half = 0; half < 2; ++half) {
                const int d0 = 32 * s + 16 * h + 8 * half;
#pragma unroll
                for (int j = 0; j < 8; ++j) {
                    float f = zr[d0 + j];
                    unsigned short hb = f2bf_u16(f);
                    Ah[s][8 * half + j] = u16_bf(hb);
                    Al[s][8 * half + j] = u16_bf(f2bf_u16(f - bf_f(hb)));
                }
            }
        }
    }

    float minv[8];
    int   mini[8];
#pragma unroll
    for (int i = 0; i < 8; ++i) { minv[i] = 3.4e38f; mini[i] = 0; }

    for (int kt = 0; kt < KCODE; kt += KT) {
        // --- Stage KT codes (hi+lo) into LDS: 4 threads per row, 64B each.
        {
            const int r = threadIdx.x >> 2;   // 0..63
            const int q = threadIdx.x & 3;    // 64B quarter
            const uint4* sh = (const uint4*)(cbhi + (size_t)(kt + r) * DDIM + q * 32);
            const uint4* sl = (const uint4*)(cblo + (size_t)(kt + r) * DDIM + q * 32);
            uint4* dh = (uint4*)(ldsHi + r * ROWB + q * 64);
            uint4* dl = (uint4*)(ldsLo + r * ROWB + q * 64);
#pragma unroll
            for (int j = 0; j < 4; ++j) { dh[j] = sh[j]; dl[j] = sl[j]; }
        }
        __syncthreads();

#pragma unroll
        for (int c8 = 0; c8 < KT / 16; ++c8) {
            // B layout: lane(h,n) holds 16 contiguous dims at 32s+16h of code row n.
            const unsigned char* bhp = ldsHi + (c8 * 16 + nm) * ROWB;
            const unsigned char* blp = ldsLo + (c8 * 16 + nm) * ROWB;
            v8f acc = {};
#pragma unroll
            for (int s = 0; s < 4; ++s) {
                const int off = (32 * s + 16 * h) * 2;
                union { v16bf v; v8bf p[2]; } bh, bl;   // two 16B-aligned LDS loads
                bh.p[0] = *(const v8bf*)(bhp + off);
                bh.p[1] = *(const v8bf*)(bhp + off + 16);
                bl.p[0] = *(const v8bf*)(blp + off);
                bl.p[1] = *(const v8bf*)(blp + off + 16);
                acc = wmma_bf16(Ah[s], bh.v, acc);      // hi*hi
                acc = wmma_bf16(Ah[s], bl.v, acc);      // hi*lo
                acc = wmma_bf16(Al[s], bh.v, acc);      // lo*hi  (~fp32-accurate dot)
            }
            const int   code = kt + c8 * 16 + nm;
            const float nrm  = cbnorm[code];
            // C layout: elem i of lane(h,n) = C[row i+8h][col n]
#pragma unroll
            for (int i = 0; i < 8; ++i) {
                float d = fmaf(-2.f, acc[i], nrm);
                if (d < minv[i]) { minv[i] = d; mini[i] = code; }
            }
        }
        __syncthreads();
    }

    // --- Reduce argmin across the 16 columns of each half-wave (tie -> lowest idx).
#pragma unroll
    for (int off = 1; off <= 8; off <<= 1) {
#pragma unroll
        for (int i = 0; i < 8; ++i) {
            float ov = __shfl_xor(minv[i], off, 32);
            int   oi = __shfl_xor(mini[i], off, 32);
            if (ov < minv[i] || (ov == minv[i] && oi < mini[i])) { minv[i] = ov; mini[i] = oi; }
        }
    }
    if (nm == 0) {
#pragma unroll
        for (int i = 0; i < 8; ++i) out_idx[row_base + 8 * h + i] = mini[i];
    }
}

// ---------------------------------------------------------------------------
// Kernel 3: gather quantized rows (== quantized_z forward), emit indices,
// per-block partial sums of (q - z)^2 (deterministic, no atomics).
// ---------------------------------------------------------------------------
__global__ __launch_bounds__(256) void vq_gather(const float* __restrict__ z,
                                                 const float* __restrict__ cb,
                                                 const int* __restrict__ idx,
                                                 float* __restrict__ out,
                                                 double* __restrict__ partial) {
    __shared__ float wsum[8];
    const int wave = threadIdx.x >> 5, lane = threadIdx.x & 31;
    const int row  = blockIdx.x * 8 + wave;
    const int code = idx[row];
    float4 q  = ((const float4*)(cb + (size_t)code * DDIM))[lane];
    float4 zz = ((const float4*)(z  + (size_t)row  * DDIM))[lane];
    ((float4*)(out + (size_t)row * DDIM))[lane] = q;   // quantized_z
    float dx = q.x - zz.x, dy = q.y - zz.y, dz = q.z - zz.z, dw = q.w - zz.w;
    float s = dx * dx + dy * dy + dz * dz + dw * dw;
#pragma unroll
    for (int off = 16; off; off >>= 1) s += __shfl_xor(s, off, 32);
    if (lane == 0) {
        out[(size_t)NROWS * DDIM + row] = (float)code;  // indices as f32
        wsum[wave] = s;
    }
    __syncthreads();
    if (threadIdx.x == 0) {
        float t = 0.f;
#pragma unroll
        for (int i = 0; i < 8; ++i) t += wsum[i];
        partial[blockIdx.x] = (double)t;
    }
}

// ---------------------------------------------------------------------------
// Kernel 4: fixed-order final reduction + scalar outputs.
// ---------------------------------------------------------------------------
__global__ __launch_bounds__(256) void vq_finalize(const double* __restrict__ partial,
                                                   float* __restrict__ out) {
    __shared__ double sh[256];
    double s = 0.0;
    for (int i = threadIdx.x; i < NROWS / 8; i += 256) s += partial[i];
    sh[threadIdx.x] = s;
    __syncthreads();
    for (int off = 128; off; off >>= 1) {
        if (threadIdx.x < (unsigned)off) sh[threadIdx.x] += sh[threadIdx.x + off];
        __syncthreads();
    }
    if (threadIdx.x == 0) {
        const double mse  = sh[0] / (double)((size_t)NROWS * DDIM);
        const size_t base = (size_t)NROWS * DDIM + NROWS;
        out[base + 0] = (float)(1.25 * mse);  // 0.25*commit + 1.0*quant
        out[base + 1] = (float)mse;           // commitment_loss
        out[base + 2] = (float)mse;           // quantization_loss
    }
}

extern "C" void kernel_launch(void* const* d_in, const int* in_sizes, int n_in,
                              void* d_out, int out_size, void* d_ws, size_t ws_size,
                              hipStream_t stream) {
    const float* z  = (const float*)d_in[0];
    const float* cb = (const float*)d_in[1];
    unsigned char* ws = (unsigned char*)d_ws;

    // Workspace layout (~2.2 MB total)
    unsigned short* cbhi = (unsigned short*)(ws);                       // 1 MB
    unsigned short* cblo = (unsigned short*)(ws + 1048576);             // 1 MB
    float*  cbnorm  = (float*)(ws + 2097152);                           // 16 KB
    int*    indices = (int*)(ws + 2113536);                             // 128 KB
    double* partial = (double*)(ws + 2244608);                          // 32 KB
    float*  out     = (float*)d_out;

    vq_cb_prep<<<KCODE / 8, 256, 0, stream>>>(cb, cbhi, cblo, cbnorm);
    vq_argmin <<<NROWS / 128, 256, 0, stream>>>(z, cbhi, cblo, cbnorm, indices);
    vq_gather <<<NROWS / 8, 256, 0, stream>>>(z, cb, indices, out, partial);
    vq_finalize<<<1, 256, 0, stream>>>(partial, out);
}
